// MultiscreenModel_69612829934242
// MI455X (gfx1250) — matrix operand order repaired and checked by reading, not compiled, not run on hardware
//
#include <hip/hip_runtime.h>
#include <math.h>

// ---------------- model constants ----------------
#define VOCAB 32000
#define DE    1024
#define DKV   64
#define NH    16
#define NL    4
#define T_SEQ 2048
#define B_SZ  2
#define BT    (B_SZ * T_SEQ)
#define WTH_F 128.0f
#define PI_F  3.14159265358979323846f

typedef __bf16 bf16;
typedef __attribute__((ext_vector_type(16))) __bf16 v16bf;
typedef __attribute__((ext_vector_type(8)))  __bf16 v8bf;
typedef __attribute__((ext_vector_type(8)))  float  v8f;
typedef __attribute__((ext_vector_type(4)))  unsigned int u32x4;
typedef __attribute__((ext_vector_type(8)))  int i32x8;
typedef __attribute__((ext_vector_type(4)))  int i32x4;

#if __has_builtin(__builtin_amdgcn_tensor_load_to_lds) && \
    __has_builtin(__builtin_amdgcn_s_wait_tensorcnt)
#define USE_TDM 1
#else
#define USE_TDM 0
#endif

// wave32 butterfly all-reduce (result in every lane)
__device__ __forceinline__ float wave_sum(float x) {
#pragma unroll
  for (int m = 16; m > 0; m >>= 1) x += __shfl_xor(x, m, 32);
  return x;
}

// combine two 16B halves into one 16-elem bf16 fragment
__device__ __forceinline__ v16bf load_frag(const bf16* p, int gap) {
  v8bf lo = *(const v8bf*)p;
  v8bf hi = *(const v8bf*)(p + gap);
  v16bf r;
#pragma unroll
  for (int i = 0; i < 8; ++i) { r[i] = lo[i]; r[8 + i] = hi[i]; }
  return r;
}

#if USE_TDM
// Issue one 2D TDM load: tile (32 k-elems x `rows`), element=2B, row stride K elems.
// g1 carries the dims; g0 built here from lds offset + global byte address.
__device__ __forceinline__ void tdm_issue(unsigned lds_addr, unsigned long long ga,
                                          const i32x8& g1) {
  u32x4 g0;
  g0[0] = 1u;                                                  // count=1
  g0[1] = lds_addr;                                            // lds_addr (bytes)
  g0[2] = (unsigned)ga;                                        // global_addr[31:0]
  g0[3] = ((unsigned)(ga >> 32) & 0x01FFFFFFu) | 0x80000000u;  // addr hi | type=2
  const i32x4 g2 = {};
  const i32x4 g3 = {};
#if __clang_major__ >= 23
  const i32x8 g4 = {};
  __builtin_amdgcn_tensor_load_to_lds(g0, g1, g2, g3, g4, 0);
#else
  __builtin_amdgcn_tensor_load_to_lds(g0, g1, g2, g3, 0);
#endif
}

__device__ __forceinline__ i32x8 tdm_desc(int rows, int K) {
  i32x8 g1 = {};
  g1[0] = 0x10000;                 // workgroup_mask=0, data_size=1 (2 bytes)
  g1[1] = 0;                       // tensor_dim0 (1<<20) low16 = 0
  g1[2] = 0x0010;                  // tensor_dim0 hi16 | tensor_dim1 low16(=0)
  g1[3] = (32 << 16) | 0x0010;     // tile_dim0=32 | tensor_dim1 hi16
  g1[4] = rows;                    // tile_dim1 = rows, tile_dim2 = 0
  g1[5] = K;                       // tensor_dim0_stride low32
  return g1;
}
#endif

// ---------------- kernel: row-normalize embedding -> bf16 ----------------
__global__ void __launch_bounds__(256) k_norm_embed(const float* __restrict__ embed,
                                                    bf16* __restrict__ Wn) {
  const int row = blockIdx.x;
  const float* src = embed + (size_t)row * DE;
  bf16* dst = Wn + (size_t)row * DE;
  const int tid = threadIdx.x;
  float v[4];
  float ss = 0.f;
#pragma unroll
  for (int i = 0; i < 4; ++i) { v[i] = src[tid + i * 256]; ss += v[i] * v[i]; }
  ss = wave_sum(ss);
  __shared__ float red[8];
  const int wid = tid >> 5, lane = tid & 31;
  if (lane == 0) red[wid] = ss;
  __syncthreads();
  float tot = red[0];
#pragma unroll
  for (int i = 1; i < 8; ++i) tot += red[i];
  const float inv = 1.0f / fmaxf(sqrtf(tot), 1e-12f);
#pragma unroll
  for (int i = 0; i < 4; ++i) dst[tid + i * 256] = (bf16)(v[i] * inv);
}

// ---------------- kernel: f32 -> bf16 convert ----------------
__global__ void k_cvt(const float* __restrict__ src, bf16* __restrict__ dst, int n) {
  int i = blockIdx.x * blockDim.x + threadIdx.x;
  const int stride = gridDim.x * blockDim.x;
  for (; i < n; i += stride) dst[i] = (bf16)src[i];
}

// ---------------- kernel: embedding gather ----------------
__global__ void k_gather(const int* __restrict__ ids, const bf16* __restrict__ Wn,
                         const float* __restrict__ sE, float* __restrict__ x) {
  const int bt = blockIdx.x;
  const int id = ids[bt];
  const float sc = expf(sE[0]);
  const bf16* src = Wn + (size_t)id * DE;
  float* dst = x + (size_t)bt * DE;
  for (int i = threadIdx.x; i < DE; i += blockDim.x) dst[i] = (float)src[i] * sc;
}

// ---------------- kernel: bf16 WMMA GEMM  C[M,N] (+)= A[M,K] * B[N,K]^T ----------------
// 256 threads = 8 waves arranged 4(M) x 2(N); block tile 128x64, wave tile 32x32
// (2x2 WMMA register blocking).  Double-buffered TDM pipeline: wave0 DMAs the B tile
// (64x32, 4KB), wave1 DMAs the A tile (128x32, 8KB) for chunk c+1 while all waves
// compute chunk c from LDS (ds_load_b128 fragments).  One barrier per chunk.
__global__ void __launch_bounds__(256) k_gemm(const bf16* __restrict__ A,
                                              const bf16* __restrict__ Bm,
                                              float* __restrict__ C,
                                              int M, int N, int K, int ldc,
                                              int accumulate,
                                              const float* __restrict__ scale_ptr) {
  const int lane = threadIdx.x & 31;
  const int wave = threadIdx.x >> 5;                  // 0..7
  const int wm = wave >> 1;                           // 0..3 (M position)
  const int wn = wave & 1;                            // 0..1 (N position)
  const int l15 = lane & 15, hi = lane >> 4;

  const int mblk = blockIdx.y * 128;
  const int nblk = blockIdx.x * 64;

  v8f acc00 = {}, acc01 = {}, acc10 = {}, acc11 = {};

#if USE_TDM
  __shared__ __align__(16) bf16 As[2][128 * 32];      // [buf][m_local][k]
  __shared__ __align__(16) bf16 Bs[2][64 * 32];       // [buf][n_local][k]

  const unsigned long long gA = (unsigned long long)(uintptr_t)(A  + (size_t)mblk * K);
  const unsigned long long gB = (unsigned long long)(uintptr_t)(Bm + (size_t)nblk * K);
  const unsigned ldsA[2] = { (unsigned)(uintptr_t)&As[0][0], (unsigned)(uintptr_t)&As[1][0] };
  const unsigned ldsB[2] = { (unsigned)(uintptr_t)&Bs[0][0], (unsigned)(uintptr_t)&Bs[1][0] };
  const i32x8 g1a = tdm_desc(128, K);
  const i32x8 g1b = tdm_desc(64, K);

  const int nch = K >> 5;
  if (wave == 0)      tdm_issue(ldsB[0], gB, g1b);
  else if (wave == 1) tdm_issue(ldsA[0], gA, g1a);

  for (int c = 0; c < nch; ++c) {
    const int cur = c & 1;
    if (wave < 2) __builtin_amdgcn_s_wait_tensorcnt((short)0);
    __syncthreads();                                  // publish buf[cur]
    if (c + 1 < nch) {                                // prefetch next chunk via TDM
      const unsigned long long go = (unsigned long long)(c + 1) * 64ull; // 32 elems * 2B
      if (wave == 0)      tdm_issue(ldsB[cur ^ 1], gB + go, g1b);
      else if (wave == 1) tdm_issue(ldsA[cur ^ 1], gA + go, g1a);
    }
    const bf16* as_ = &As[cur][0];
    const bf16* bs_ = &Bs[cur][0];
    // A frag: lane=row; lower half-wave K {0..7,16..23}, upper {8..15,24..31}
    v16bf a0 = load_frag(as_ + (wm * 32 + l15) * 32 + hi * 8, 16);
    v16bf a1 = load_frag(as_ + (wm * 32 + 16 + l15) * 32 + hi * 8, 16);
    // B frag: lane=column; lower half-wave K {0..7,16..23}, upper {8..15,24..31}
    v16bf b0 = load_frag(bs_ + (wn * 32 + l15) * 32 + hi * 8, 16);
    v16bf b1 = load_frag(bs_ + (wn * 32 + 16 + l15) * 32 + hi * 8, 16);

    acc00 = __builtin_amdgcn_wmma_f32_16x16x32_bf16(false, a0, false, b0, (short)0, acc00, false, false);
    acc01 = __builtin_amdgcn_wmma_f32_16x16x32_bf16(false, a0, false, b1, (short)0, acc01, false, false);
    acc10 = __builtin_amdgcn_wmma_f32_16x16x32_bf16(false, a1, false, b0, (short)0, acc10, false, false);
    acc11 = __builtin_amdgcn_wmma_f32_16x16x32_bf16(false, a1, false, b1, (short)0, acc11, false, false);
  }
#else
  const bf16* arow0 = A + (size_t)(mblk + wm * 32 + l15) * K + hi * 8;
  const bf16* arow1 = arow0 + (size_t)16 * K;
  const bf16* brow0 = Bm + (size_t)(nblk + wn * 32 + l15) * K + hi * 8;
  const bf16* brow1 = brow0 + (size_t)16 * K;
  for (int k0 = 0; k0 < K; k0 += 32) {
    v16bf a0 = load_frag(arow0 + k0, 16);
    v16bf a1 = load_frag(arow1 + k0, 16);
    v16bf b0 = load_frag(brow0 + k0, 16);
    v16bf b1 = load_frag(brow1 + k0, 16);
    acc00 = __builtin_amdgcn_wmma_f32_16x16x32_bf16(false, a0, false, b0, (short)0, acc00, false, false);
    acc01 = __builtin_amdgcn_wmma_f32_16x16x32_bf16(false, a0, false, b1, (short)0, acc01, false, false);
    acc10 = __builtin_amdgcn_wmma_f32_16x16x32_bf16(false, a1, false, b0, (short)0, acc10, false, false);
    acc11 = __builtin_amdgcn_wmma_f32_16x16x32_bf16(false, a1, false, b1, (short)0, acc11, false, false);
  }
#endif

  const float sc = scale_ptr ? expf(scale_ptr[0]) : 1.0f;
  const v8f* accs[4] = { &acc00, &acc01, &acc10, &acc11 };
#pragma unroll
  for (int si = 0; si < 2; ++si) {
#pragma unroll
    for (int sj = 0; sj < 2; ++sj) {
      const v8f a = *accs[si * 2 + sj];
      const int cm = mblk + wm * 32 + si * 16 + hi * 8;
      const int cn = nblk + wn * 32 + sj * 16 + l15;
      float* cp = C + (size_t)cm * ldc + cn;
#pragma unroll
      for (int i = 0; i < 8; ++i) {
        const float v = a[i] * sc;
        if (accumulate) cp[(size_t)i * ldc] += v;
        else            cp[(size_t)i * ldc]  = v;
      }
    }
  }
}

// ---------------- kernel: per-head normalize q,k,v + MIPE rotate (in place) ----------------
__global__ void __launch_bounds__(512) k_prep(float* __restrict__ qkvg,
                                              const float* __restrict__ sw) {
  const int bt = blockIdx.x;
  const int t = bt % T_SEQ;
  const int head = threadIdx.x >> 5;
  const int lane = threadIdx.x & 31;
  float* base = qkvg + (size_t)bt * (4 * DE);
  const float w = expf(sw[head]) + 1.0f;
  const float phi = (w < WTH_F) ? 0.5f * (cosf(PI_F * w / WTH_F) + 1.0f) : 0.0f;
  const float ang = (float)t * (phi / w);
  const float ca = cosf(ang), sa = sinf(ang);
#pragma unroll
  for (int part = 0; part < 3; ++part) {        // q, k, v
    float* p = base + part * DE + head * DKV + lane * 2;
    float x0 = p[0], x1 = p[1];
    const float n2 = wave_sum(x0 * x0 + x1 * x1);
    const float inv = 1.0f / fmaxf(sqrtf(n2), 1e-12f);
    x0 *= inv; x1 *= inv;
    if (part < 2 && lane == 0) {                // MIPE rotates dims 0,1 of q and k
      const float r0 = x0 * ca - x1 * sa;
      const float r1 = x0 * sa + x1 * ca;
      x0 = r0; x1 = r1;
    }
    p[0] = x0; p[1] = x1;
  }
}

// ---------------- kernel: banded attention + gate ----------------
__global__ void __launch_bounds__(512) k_attn(const float* __restrict__ qkvg,
                                              float* __restrict__ h,
                                              const float* __restrict__ sw,
                                              const float* __restrict__ sr,
                                              const float* __restrict__ sO) {
  const int bt = blockIdx.x;
  const int t = bt % T_SEQ;
  const int b = bt / T_SEQ;
  const int head = threadIdx.x >> 5;
  const int lane = threadIdx.x & 31;
  const float* base = qkvg + (size_t)bt * (4 * DE);

  const float q0 = base[head * DKV + lane * 2 + 0];
  const float q1 = base[head * DKV + lane * 2 + 1];
  const float w = expf(sw[head]) + 1.0f;
  const float r = expf(sr[head]) + 1.0f;
  const float winv = 1.0f / w;

  int s_lo = t - (int)ceilf(w);
  if (s_lo < 0) s_lo = 0;
  float a0 = 0.f, a1 = 0.f;
  for (int s = s_lo; s <= t; ++s) {
    const float dt = (float)(t - s);
    if (dt >= w) continue;                         // strict: rel > -w
    const size_t rowoff = (size_t)(b * T_SEQ + s) * (4 * DE) + head * DKV + lane * 2;
    const float k0 = qkvg[rowoff + DE + 0];
    const float k1 = qkvg[rowoff + DE + 1];
    const float sim = wave_sum(q0 * k0 + q1 * k1);
    float z = 1.0f - r * (1.0f - sim);
    z = fmaxf(z, 0.0f);
    const float mask = 0.5f * (cosf(PI_F * dt * winv) + 1.0f);
    const float rho = z * z * mask;
    a0 += rho * qkvg[rowoff + 2 * DE + 0];
    a1 += rho * qkvg[rowoff + 2 * DE + 1];
  }
  const float hn2 = wave_sum(a0 * a0 + a1 * a1);
  const float hn = fmaxf(sqrtf(hn2), 1e-8f);
  const float usc = tanhf(hn) / hn;
  const float g0 = base[3 * DE + head * DKV + lane * 2 + 0];
  const float g1 = base[3 * DE + head * DKV + lane * 2 + 1];
  const float gh0 = tanhf(g0 / (1.0f + expf(-g0)));   // tanh(silu(g))
  const float gh1 = tanhf(g1 / (1.0f + expf(-g1)));
  const float so = expf(sO[head]);
  float* dst = h + (size_t)bt * DE + head * DKV + lane * 2;
  dst[0] = usc * a0 * gh0 * so;
  dst[1] = usc * a1 * gh1 * so;
}

// ---------------- launcher ----------------
extern "C" void kernel_launch(void* const* d_in, const int* in_sizes, int n_in,
                              void* d_out, int out_size, void* d_ws, size_t ws_size,
                              hipStream_t stream) {
  (void)in_sizes; (void)n_in; (void)out_size; (void)ws_size;
  const int*   ids = (const int*)d_in[0];
  const float* emb = (const float*)d_in[1];
  const float* sE  = (const float*)d_in[2];
  const float* sF  = (const float*)d_in[3];
  const float* Wq  = (const float*)d_in[4];
  const float* Wk  = (const float*)d_in[5];
  const float* Wv  = (const float*)d_in[6];
  const float* Wg  = (const float*)d_in[7];
  const float* Wo  = (const float*)d_in[8];
  const float* sw  = (const float*)d_in[9];
  const float* sr  = (const float*)d_in[10];
  const float* sO  = (const float*)d_in[11];
  float* out = (float*)d_out;

  char* ws = (char*)d_ws;
  size_t off = 0;
  bf16*  Wn     = (bf16*)(void*)(ws + off);  off += (size_t)VOCAB * DE * 2;
  float* x      = (float*)(void*)(ws + off); off += (size_t)BT * DE * 4;
  bf16*  xb     = (bf16*)(void*)(ws + off);  off += (size_t)BT * DE * 2;
  bf16*  wqkvgb = (bf16*)(void*)(ws + off);  off += (size_t)NL * 4 * DE * DE * 2; // [l][q|k|v|g]
  bf16*  wob    = (bf16*)(void*)(ws + off);  off += (size_t)NL * DE * DE * 2;
  float* qkvg   = (float*)(void*)(ws + off); off += (size_t)BT * 4 * DE * 4;
  float* hbuf   = (float*)(void*)(ws + off); off += (size_t)BT * DE * 4;
  bf16*  hb     = (bf16*)(void*)(ws + off);  off += (size_t)BT * DE * 2;

  const int WSZ = DE * DE;

  k_norm_embed<<<VOCAB, 256, 0, stream>>>(emb, Wn);
  // pack per-layer [Wq;Wk;Wv;Wg] (bf16) and Wo (bf16)
  for (int l = 0; l < NL; ++l) {
    bf16* dst = wqkvgb + (size_t)l * 4 * WSZ;
    k_cvt<<<1024, 256, 0, stream>>>(Wq + (size_t)l * WSZ, dst + 0 * WSZ, WSZ);
    k_cvt<<<1024, 256, 0, stream>>>(Wk + (size_t)l * WSZ, dst + 1 * WSZ, WSZ);
    k_cvt<<<1024, 256, 0, stream>>>(Wv + (size_t)l * WSZ, dst + 2 * WSZ, WSZ);
    k_cvt<<<1024, 256, 0, stream>>>(Wg + (size_t)l * WSZ, dst + 3 * WSZ, WSZ);
  }
  k_cvt<<<2048, 256, 0, stream>>>(Wo, wob, NL * WSZ);
  k_gather<<<BT, 256, 0, stream>>>(ids, Wn, sE, x);

  const dim3 gQ(4 * DE / 64, BT / 128);    // 64 x 32   fused QKVG GEMM
  const dim3 gO(DE / 64, BT / 128);        // 16 x 32   Wo GEMM
  const dim3 gL(VOCAB / 64, BT / 128);     // 500 x 32  logits GEMM

  for (int l = 0; l < NL; ++l) {
    k_cvt<<<2048, 256, 0, stream>>>(x, xb, BT * DE);
    k_gemm<<<gQ, 256, 0, stream>>>(xb, wqkvgb + (size_t)l * 4 * WSZ, qkvg,
                                   BT, 4 * DE, DE, 4 * DE, 0, nullptr);
    k_prep<<<BT, 512, 0, stream>>>(qkvg, sw + l * NH);
    k_attn<<<BT, 512, 0, stream>>>(qkvg, hbuf, sw + l * NH, sr + l * NH, sO + l * NH);
    k_cvt<<<2048, 256, 0, stream>>>(hbuf, hb, BT * DE);
    k_gemm<<<gO, 256, 0, stream>>>(hb, wob + (size_t)l * WSZ, x,
                                   BT, DE, DE, DE, 1, nullptr);   // residual add
  }

  k_cvt<<<2048, 256, 0, stream>>>(x, xb, BT * DE);
  k_gemm<<<gL, 256, 0, stream>>>(xb, Wn, out, BT, VOCAB, DE, VOCAB, 0, sF);
}
